// CoAttNet_26680336843075
// MI455X (gfx1250) — compile-verified, matrix operand order checked by reading, not run on hardware
//
#include <hip/hip_runtime.h>
#include <hip/hip_bf16.h>
#include <math.h>

typedef __bf16 bf16t;
typedef __attribute__((ext_vector_type(16))) __bf16 v16bf;
typedef __attribute__((ext_vector_type(8)))  __bf16 v8bf;
typedef __attribute__((ext_vector_type(4)))  __bf16 v4bf;
typedef __attribute__((ext_vector_type(8)))  float  v8f;
typedef __attribute__((ext_vector_type(4)))  int    v4i;

// Problem constants (match reference setup_inputs)
#define BB 4
#define CC 256
#define AA 4096   // H*W

// GEMM tiling: 256 threads = 8 wave32s, 128x128 block tile, each wave 32x64 (2x4 WMMA)
#define TM   128
#define TN   128
#define TKC  32
#define NTHR 256
#define LDSW 40   // padded row stride (80B: 16B-aligned halves, 20-bank stride)

// ---- CDNA5 async global->LDS path (guarded; falls back to register double-buffer) ----
#if defined(__has_builtin)
# if __has_builtin(__builtin_amdgcn_global_load_async_to_lds_b128) && \
     __has_builtin(__builtin_amdgcn_s_wait_asynccnt)
#  define USE_ASYNC_LDS 1
# endif
#endif
#ifndef USE_ASYNC_LDS
# define USE_ASYNC_LDS 0
#endif

__device__ __forceinline__ void async_ld_b128(const void* g, void* l) {
#if USE_ASYNC_LDS
    auto gp = (__attribute__((address_space(1))) v4i*)(unsigned long long)(uintptr_t)g;
    auto lp = (__attribute__((address_space(3))) v4i*)(unsigned)(uintptr_t)l;
    __builtin_amdgcn_global_load_async_to_lds_b128(gp, lp, 0, 0);
#else
    (void)g; (void)l;
#endif
}

// k-group permutation so each lane's 16 fragment elements are contiguous in LDS:
// element e, lane-half h wants k=((e<8)?e:e+8)+8h  ->  swap k-groups 1 and 2.
__device__ __forceinline__ constexpr int kperm(int k) {
    return ((k + 8) & 16) ? (k ^ 24) : k;
}

// 32B contiguous fragment load = 2 x ds_load_b128
__device__ __forceinline__ v16bf frag_ld(const bf16t* p) {
    const v8bf* q = (const v8bf*)p;
    v8bf lo = q[0], hi = q[1];
    return __builtin_shufflevector(lo, hi, 0, 1, 2, 3, 4, 5, 6, 7,
                                   8, 9, 10, 11, 12, 13, 14, 15);
}

#define WMMA_BF16(a, b, c) \
    __builtin_amdgcn_wmma_f32_16x16x32_bf16(false, (a), false, (b), (short)0, (c), false, false)

// one k-step: 2 A-frags x 4 B-frags -> 8 WMMAs per wave
__device__ __forceinline__ void mma_step(const bf16t* lA, const bf16t* lB,
                                         v8f (&acc)[2][4], int wr, int wc) {
    const int lane = (int)(threadIdx.x & 31u);
    const int half16 = (lane >> 4) << 4;
    const int lr = lane & 15;
    v16bf a[2], b[4];
#pragma unroll
    for (int i = 0; i < 2; ++i)
        a[i] = frag_ld(&lA[(wr * 32 + i * 16 + lr) * LDSW + half16]);
#pragma unroll
    for (int j = 0; j < 4; ++j)
        b[j] = frag_ld(&lB[(wc * 64 + j * 16 + lr) * LDSW + half16]);
#pragma unroll
    for (int i = 0; i < 2; ++i)
#pragma unroll
        for (int j = 0; j < 4; ++j)
            acc[i][j] = WMMA_BF16(a[i], b[j], acc[i][j]);
}

// D 16x16 f32 layout: VGPR r, lane l -> M = r + 8*(l>=16), N = l%16
__device__ __forceinline__ void store_acc_f32(float* D, int m0, int n0, int wr, int wc,
                                              v8f (&acc)[2][4]) {
    const int lane = (int)(threadIdx.x & 31u);
    const int ncol = lane & 15, mh = (lane >> 4) << 3;
#pragma unroll
    for (int i = 0; i < 2; ++i)
#pragma unroll
        for (int j = 0; j < 4; ++j) {
            int gm = m0 + wr * 32 + i * 16 + mh;
            int gn = n0 + wc * 64 + j * 16 + ncol;
#pragma unroll
            for (int r = 0; r < 8; ++r)
                D[(size_t)(gm + r) * AA + gn] = acc[i][j][r];
        }
}

__device__ __forceinline__ void store_acc_bf16(bf16t* D, int m0, int n0, int wr, int wc,
                                               v8f (&acc)[2][4]) {
    const int lane = (int)(threadIdx.x & 31u);
    const int ncol = lane & 15, mh = (lane >> 4) << 3;
#pragma unroll
    for (int i = 0; i < 2; ++i)
#pragma unroll
        for (int j = 0; j < 4; ++j) {
            int gm = m0 + wr * 32 + i * 16 + mh;
            int gn = n0 + wc * 64 + j * 16 + ncol;
#pragma unroll
            for (int r = 0; r < 8; ++r)
                D[(size_t)(gm + r) * AA + gn] = (bf16t)acc[i][j][r];
        }
}

// ---------------- staging helpers (fixed chunk map, no branchy loops) ----------------
// A tile 128x32 bf16: 512 x 16B chunks -> 2 per thread. chunk c: row=c>>2, kgroup=(c&3)*8
template<int LDA>
__device__ __forceinline__ void loadA_regs(const bf16t* __restrict__ Ag, int t, int m0, int k0,
                                           v8bf (&pa)[2]) {
#pragma unroll
    for (int j = 0; j < 2; ++j) {
        int c = t + j * NTHR, r = c >> 2, kb = (c & 3) * 8;
        pa[j] = *(const v8bf*)&Ag[(size_t)(m0 + r) * LDA + (k0 + kb)];
    }
}
__device__ __forceinline__ void storeA_regs(bf16t* lbuf, int t, const v8bf (&pa)[2]) {
#pragma unroll
    for (int j = 0; j < 2; ++j) {
        int c = t + j * NTHR, r = c >> 2, kb = (c & 3) * 8;
        *(v8bf*)&lbuf[r * LDSW + kperm(kb)] = pa[j];
    }
}
template<int LDA>
__device__ __forceinline__ void stageA_async(const bf16t* __restrict__ Ag, bf16t* lbuf,
                                             int t, int m0, int k0) {
#pragma unroll
    for (int j = 0; j < 2; ++j) {
        int c = t + j * NTHR, r = c >> 2, kb = (c & 3) * 8;
        async_ld_b128(&Ag[(size_t)(m0 + r) * LDA + (k0 + kb)],
                      &lbuf[r * LDSW + kperm(kb)]);
    }
}
// B tile 32x128 bf16 from [k][n] global: 1024 x 8B chunks -> 4/thread. c: k=c>>5, nb=(c&31)*4
__device__ __forceinline__ void loadB_bf16(const bf16t* __restrict__ Bg, int t, int n0, int k0,
                                           v4bf (&pb)[4]) {
#pragma unroll
    for (int j = 0; j < 4; ++j) {
        int c = t + j * NTHR, k = c >> 5, nb = (c & 31) * 4;
        pb[j] = *(const v4bf*)&Bg[(size_t)(k0 + k) * AA + (n0 + nb)];
    }
}
__device__ __forceinline__ void storeB_bf16(bf16t* lB, int t, const v4bf (&pb)[4]) {
#pragma unroll
    for (int j = 0; j < 4; ++j) {
        int c = t + j * NTHR, k = c >> 5, nb = (c & 31) * 4;
        int pos = kperm(k);
#pragma unroll
        for (int e = 0; e < 4; ++e) lB[(nb + e) * LDSW + pos] = pb[j][e];
    }
}

// ---------------- elementwise converts ----------------

__global__ __launch_bounds__(256) void k_f32_to_bf16(const float* __restrict__ src,
                                                     bf16t* __restrict__ dst, int n) {
    int i = blockIdx.x * 256 + threadIdx.x;
    if (i < n) dst[i] = (bf16t)src[i];
}

// ---- VaW = W(256x256,row-major) @ Va(256x4096) -> bf16 [256][4096] ----

__global__ __launch_bounds__(NTHR) void k_gemm_nn_bf16(const bf16t* __restrict__ Ag,
                                                       const bf16t* __restrict__ Bg,
                                                       bf16t* __restrict__ Dg) {
    __shared__ bf16t lA[2][TM * LDSW];
    __shared__ bf16t lB[TN * LDSW];
    const int t = (int)threadIdx.x;
    const int m0 = blockIdx.y * TM, n0 = blockIdx.x * TN;
    const int w = t >> 5, wr = w >> 1, wc = w & 1;
    v8f acc[2][4] = {};

    v4bf pb[4];
    loadB_bf16(Bg, t, n0, 0, pb);
#if USE_ASYNC_LDS
    stageA_async<CC>(Ag, lA[0], t, m0, 0);
#else
    v8bf pa[2];
    loadA_regs<CC>(Ag, t, m0, 0, pa);
#endif
    int buf = 0;
    for (int k0 = 0; k0 < CC; k0 += TKC) {
        const bool more = (k0 + TKC) < CC;
#if USE_ASYNC_LDS
        if (more) {
            stageA_async<CC>(Ag, lA[buf ^ 1], t, m0, k0 + TKC);
            __builtin_amdgcn_s_wait_asynccnt(2);   // current tile landed; next 2 in flight
        } else {
            __builtin_amdgcn_s_wait_asynccnt(0);
        }
#else
        storeA_regs(lA[buf], t, pa);
        if (more) loadA_regs<CC>(Ag, t, m0, k0 + TKC, pa);
#endif
        storeB_bf16(lB, t, pb);
        if (more) loadB_bf16(Bg, t, n0, k0 + TKC, pb);
        __syncthreads();
        mma_step(lA[buf], lB, acc, wr, wc);
        buf ^= 1;
        __syncthreads();
    }
    store_acc_bf16(Dg, m0, n0, wr, wc, acc);
}

// ---- S[a][n] = sum_c VaW[c][a] * Vb[c][n]  (A transposed-in-memory, f32 out) ----

__global__ __launch_bounds__(NTHR) void k_gemm_tn_f32(const bf16t* __restrict__ Ag,
                                                      const bf16t* __restrict__ Bg,
                                                      float* __restrict__ Dg) {
    __shared__ bf16t lA[TM * LDSW];
    __shared__ bf16t lB[TN * LDSW];
    const int t = (int)threadIdx.x;
    const int m0 = blockIdx.y * TM, n0 = blockIdx.x * TN;
    const int w = t >> 5, wr = w >> 1, wc = w & 1;
    v8f acc[2][4] = {};

    // A stored [k][m]: 512 x 16B chunks -> 2/thread. c: k=c>>4, mb=(c&15)*8 (scatter rows)
    v8bf pa[2];
    v4bf pb[4];
#pragma unroll
    for (int j = 0; j < 2; ++j) {
        int c = t + j * NTHR, k = c >> 4, mb = (c & 15) * 8;
        pa[j] = *(const v8bf*)&Ag[(size_t)k * AA + (m0 + mb)];
    }
    loadB_bf16(Bg, t, n0, 0, pb);

    for (int k0 = 0; k0 < CC; k0 += TKC) {
        const bool more = (k0 + TKC) < CC;
#pragma unroll
        for (int j = 0; j < 2; ++j) {
            int c = t + j * NTHR, k = c >> 4, mb = (c & 15) * 8;
            int pos = kperm(k);
#pragma unroll
            for (int e = 0; e < 8; ++e) lA[(mb + e) * LDSW + pos] = pa[j][e];
        }
        storeB_bf16(lB, t, pb);
        if (more) {
#pragma unroll
            for (int j = 0; j < 2; ++j) {
                int c = t + j * NTHR, k = c >> 4, mb = (c & 15) * 8;
                pa[j] = *(const v8bf*)&Ag[(size_t)(k0 + TKC + k) * AA + (m0 + mb)];
            }
            loadB_bf16(Bg, t, n0, k0 + TKC, pb);
        }
        __syncthreads();
        mma_step(lA, lB, acc, wr, wc);
        __syncthreads();
    }
    store_acc_f32(Dg, m0, n0, wr, wc, acc);
}

// ---------------- softmax statistics (single pass over L2-resident S) ----------------

__device__ __forceinline__ void online_merge(float& m, float& s, float m2, float s2) {
    float mn = fmaxf(m, m2);
    s = s * __expf(m - mn) + s2 * __expf(m2 - mn);
    m = mn;
}

__global__ __launch_bounds__(256) void k_row_stats(const float* __restrict__ S,
                                                   float* __restrict__ rmax,
                                                   float* __restrict__ rsum) {
    __shared__ float sm[256], ss[256];
    const int a = blockIdx.x;
    const float* row = S + (size_t)a * AA;
    float m = -3.0e38f, s = 0.f;
    for (int n = (int)threadIdx.x; n < AA; n += 256) {
        float v = row[n];
        float mn = fmaxf(m, v);
        s = s * __expf(m - mn) + __expf(v - mn);
        m = mn;
    }
    sm[threadIdx.x] = m; ss[threadIdx.x] = s;
    __syncthreads();
    for (int off = 128; off > 0; off >>= 1) {
        if ((int)threadIdx.x < off) {
            float m1 = sm[threadIdx.x], s1 = ss[threadIdx.x];
            online_merge(m1, s1, sm[threadIdx.x + off], ss[threadIdx.x + off]);
            sm[threadIdx.x] = m1; ss[threadIdx.x] = s1;
        }
        __syncthreads();
    }
    if (threadIdx.x == 0) { rmax[a] = sm[0]; rsum[a] = ss[0]; }
}

__global__ __launch_bounds__(256) void k_col_stats(const float* __restrict__ S,
                                                   float* __restrict__ cmax,
                                                   float* __restrict__ csum) {
    const int n = blockIdx.x * 256 + threadIdx.x;   // column per thread, coalesced rows
    float m = -3.0e38f, s = 0.f;
    for (int a = 0; a < AA; ++a) {
        float v = S[(size_t)a * AA + n];
        float mn = fmaxf(m, v);
        s = s * __expf(m - mn) + __expf(v - mn);
        m = mn;
    }
    cmax[n] = m; csum[n] = s;
}

// ---- Va_att[c][n] = sum_a Vb16[c][a] * exp(S[a][n]-cmax[n])/csum[n] ----

__global__ __launch_bounds__(NTHR) void k_gemm_att_col(const bf16t* __restrict__ Ag,
                                                       const float* __restrict__ S,
                                                       const float* __restrict__ cmax,
                                                       const float* __restrict__ csum,
                                                       float* __restrict__ Dg) {
    __shared__ bf16t lA[2][TM * LDSW];
    __shared__ bf16t lB[TN * LDSW];
    __shared__ float s_m[TN], s_r[TN];
    const int t = (int)threadIdx.x;
    const int m0 = blockIdx.y * TM, n0 = blockIdx.x * TN;
    const int w = t >> 5, wr = w >> 1, wc = w & 1;

    if (t < TN) {
        s_m[t] = cmax[n0 + t];
        s_r[t] = 1.0f / csum[n0 + t];
    }

    v8f acc[2][4] = {};
    // B from S[k][n] f32: 1024 x 16B chunks -> 4/thread. c: k=c>>5, nb=(c&31)*4
    float4 pf[4];
#pragma unroll
    for (int j = 0; j < 4; ++j) {
        int c = t + j * NTHR, k = c >> 5, nb = (c & 31) * 4;
        pf[j] = *(const float4*)&S[(size_t)k * AA + (n0 + nb)];
    }
#if USE_ASYNC_LDS
    stageA_async<AA>(Ag, lA[0], t, m0, 0);
#else
    v8bf pa[2];
    loadA_regs<AA>(Ag, t, m0, 0, pa);
#endif
    __syncthreads();   // s_m/s_r visible
    int buf = 0;
    for (int k0 = 0; k0 < AA; k0 += TKC) {
        const bool more = (k0 + TKC) < AA;
#if USE_ASYNC_LDS
        if (more) {
            stageA_async<AA>(Ag, lA[buf ^ 1], t, m0, k0 + TKC);
            __builtin_amdgcn_s_wait_asynccnt(2);
        } else {
            __builtin_amdgcn_s_wait_asynccnt(0);
        }
#else
        storeA_regs(lA[buf], t, pa);
        if (more) loadA_regs<AA>(Ag, t, m0, k0 + TKC, pa);
#endif
#pragma unroll
        for (int j = 0; j < 4; ++j) {
            int c = t + j * NTHR, k = c >> 5, nb = (c & 31) * 4;
            int pos = kperm(k);
            lB[(nb + 0) * LDSW + pos] = (bf16t)(__expf(pf[j].x - s_m[nb + 0]) * s_r[nb + 0]);
            lB[(nb + 1) * LDSW + pos] = (bf16t)(__expf(pf[j].y - s_m[nb + 1]) * s_r[nb + 1]);
            lB[(nb + 2) * LDSW + pos] = (bf16t)(__expf(pf[j].z - s_m[nb + 2]) * s_r[nb + 2]);
            lB[(nb + 3) * LDSW + pos] = (bf16t)(__expf(pf[j].w - s_m[nb + 3]) * s_r[nb + 3]);
        }
        if (more) {
#pragma unroll
            for (int j = 0; j < 4; ++j) {
                int c = t + j * NTHR, k = c >> 5, nb = (c & 31) * 4;
                pf[j] = *(const float4*)&S[(size_t)(k0 + TKC + k) * AA + (n0 + nb)];
            }
        }
        __syncthreads();
        mma_step(lA[buf], lB, acc, wr, wc);
        buf ^= 1;
        __syncthreads();
    }
    store_acc_f32(Dg, m0, n0, wr, wc, acc);
}

// ---- Vb_att[c][n] = sum_a Va16[c][a] * exp(S[n][a]-rmax[n])/rsum[n]  (S transposed) ----

__global__ __launch_bounds__(NTHR) void k_gemm_att_row(const bf16t* __restrict__ Ag,
                                                       const float* __restrict__ S,
                                                       const float* __restrict__ rmax,
                                                       const float* __restrict__ rsum,
                                                       float* __restrict__ Dg) {
    __shared__ bf16t lA[2][TM * LDSW];
    __shared__ bf16t lB[TN * LDSW];
    __shared__ float s_m[TN], s_r[TN];
    const int t = (int)threadIdx.x;
    const int m0 = blockIdx.y * TM, n0 = blockIdx.x * TN;
    const int w = t >> 5, wr = w >> 1, wc = w & 1;

    if (t < TN) {
        s_m[t] = rmax[n0 + t];
        s_r[t] = 1.0f / rsum[n0 + t];
    }

    v8f acc[2][4] = {};
    // B from S[n][k] f32 (coalesced along k): c: n=c>>3, kb=(c&7)*4
    float4 pf[4];
#pragma unroll
    for (int j = 0; j < 4; ++j) {
        int c = t + j * NTHR, n = c >> 3, kb = (c & 7) * 4;
        pf[j] = *(const float4*)&S[(size_t)(n0 + n) * AA + kb];
    }
#if USE_ASYNC_LDS
    stageA_async<AA>(Ag, lA[0], t, m0, 0);
#else
    v8bf pa[2];
    loadA_regs<AA>(Ag, t, m0, 0, pa);
#endif
    __syncthreads();
    int buf = 0;
    for (int k0 = 0; k0 < AA; k0 += TKC) {
        const bool more = (k0 + TKC) < AA;
#if USE_ASYNC_LDS
        if (more) {
            stageA_async<AA>(Ag, lA[buf ^ 1], t, m0, k0 + TKC);
            __builtin_amdgcn_s_wait_asynccnt(2);
        } else {
            __builtin_amdgcn_s_wait_asynccnt(0);
        }
#else
        storeA_regs(lA[buf], t, pa);
        if (more) loadA_regs<AA>(Ag, t, m0, k0 + TKC, pa);
#endif
#pragma unroll
        for (int j = 0; j < 4; ++j) {
            int c = t + j * NTHR, n = c >> 3, kb = (c & 7) * 4;
            float sm = s_m[n], sr = s_r[n];
            int pos = kperm(kb);   // contiguous for kb..kb+3
            lB[n * LDSW + pos + 0] = (bf16t)(__expf(pf[j].x - sm) * sr);
            lB[n * LDSW + pos + 1] = (bf16t)(__expf(pf[j].y - sm) * sr);
            lB[n * LDSW + pos + 2] = (bf16t)(__expf(pf[j].z - sm) * sr);
            lB[n * LDSW + pos + 3] = (bf16t)(__expf(pf[j].w - sm) * sr);
        }
        if (more) {
#pragma unroll
            for (int j = 0; j < 4; ++j) {
                int c = t + j * NTHR, n = c >> 3, kb = (c & 7) * 4;
                pf[j] = *(const float4*)&S[(size_t)(n0 + n) * AA + (k0 + TKC + kb)];
            }
        }
        __syncthreads();
        mma_step(lA[buf], lB, acc, wr, wc);
        buf ^= 1;
        __syncthreads();
    }
    store_acc_f32(Dg, m0, n0, wr, wc, acc);
}

// ---------------- gate conv + sigmoid mask + concat epilogue ----------------

__global__ __launch_bounds__(256) void k_gate_concat(const float* __restrict__ att,
                                                     const float* __restrict__ inp,
                                                     const float* __restrict__ gate,
                                                     float* __restrict__ out) {
    __shared__ float g[CC];
    const int t = (int)threadIdx.x;
    if (t < CC) g[t] = gate[t];
    __syncthreads();
    const int n = blockIdx.x * 256 + t;
    float dot = 0.f;
    for (int c = 0; c < CC; ++c) dot += att[(size_t)c * AA + n] * g[c];
    float mask = 1.0f / (1.0f + __expf(-dot));
    for (int c = 0; c < CC; ++c) {
        out[(size_t)c * AA + n]        = att[(size_t)c * AA + n] * mask;
        out[(size_t)(CC + c) * AA + n] = inp[(size_t)c * AA + n];
    }
}

__global__ void k_write_scalar(const int* __restrict__ v, float* __restrict__ dst) {
    if (threadIdx.x == 0 && blockIdx.x == 0) dst[0] = (float)v[0];
}

// ---------------- host launcher ----------------

extern "C" void kernel_launch(void* const* d_in, const int* in_sizes, int n_in,
                              void* d_out, int out_size, void* d_ws, size_t ws_size,
                              hipStream_t stream) {
    (void)in_sizes; (void)n_in; (void)out_size; (void)ws_size;
    const float* Va = (const float*)d_in[0];        // [B, C, A]
    const float* Vb = (const float*)d_in[1];
    const float* Wl = (const float*)d_in[2];        // [C, C]
    const float* Wg = (const float*)d_in[3];        // [C]
    const int*   isz = (const int*)d_in[4];
    float* out = (float*)d_out;

    // workspace carve-up (reused per batch so the 64MB S stays L2-resident)
    char* ws = (char*)d_ws;
    size_t off = 0;
    auto carve = [&](size_t bytes) -> char* {
        off = (off + 255) & ~(size_t)255;
        char* p = ws + off;
        off += bytes;
        return p;
    };
    bf16t* sW   = (bf16t*)carve((size_t)CC * CC * sizeof(bf16t));
    bf16t* sVa  = (bf16t*)carve((size_t)CC * AA * sizeof(bf16t));
    bf16t* sVb  = (bf16t*)carve((size_t)CC * AA * sizeof(bf16t));
    bf16t* sVaW = (bf16t*)carve((size_t)CC * AA * sizeof(bf16t));
    float* S    = (float*)carve((size_t)AA * AA * sizeof(float));
    float* rmax = (float*)carve((size_t)AA * sizeof(float));
    float* rsum = (float*)carve((size_t)AA * sizeof(float));
    float* cmax = (float*)carve((size_t)AA * sizeof(float));
    float* csum = (float*)carve((size_t)AA * sizeof(float));
    float* attA = (float*)carve((size_t)CC * AA * sizeof(float));
    float* attB = (float*)carve((size_t)CC * AA * sizeof(float));

    const int CA = CC * AA;
    dim3 blk(NTHR), blk256(256);

    k_f32_to_bf16<<<(CC * CC + 255) / 256, blk256, 0, stream>>>(Wl, sW, CC * CC);

    for (int b = 0; b < BB; ++b) {
        const float* Vab = Va + (size_t)b * CA;
        const float* Vbb = Vb + (size_t)b * CA;

        k_f32_to_bf16<<<(CA + 255) / 256, blk256, 0, stream>>>(Vab, sVa, CA);
        k_f32_to_bf16<<<(CA + 255) / 256, blk256, 0, stream>>>(Vbb, sVb, CA);

        // VaW[c_out][a] = sum_c W[c_out][c] * Va[c][a]   (M=256, N=4096, K=256)
        k_gemm_nn_bf16<<<dim3(AA / TN, CC / TM), blk, 0, stream>>>(sW, sVa, sVaW);

        // S[a][n] = sum_c VaW[c][a] * Vb[c][n]           (M=N=4096, K=256)
        k_gemm_tn_f32<<<dim3(AA / TN, AA / TM), blk, 0, stream>>>(sVaW, sVb, S);

        k_row_stats<<<AA, blk256, 0, stream>>>(S, rmax, rsum);
        k_col_stats<<<AA / 256, blk256, 0, stream>>>(S, cmax, csum);

        // Va_att = Vb @ softmax_col(S);  Vb_att = Va @ softmax_row(S)^T
        k_gemm_att_col<<<dim3(AA / TN, CC / TM), blk, 0, stream>>>(sVb, S, cmax, csum, attA);
        k_gemm_att_row<<<dim3(AA / TN, CC / TM), blk, 0, stream>>>(sVa, S, rmax, rsum, attB);

        float* outA = out + (size_t)b * 2 * CA;
        float* outB = out + (size_t)BB * 2 * CA + (size_t)b * 2 * CA;
        k_gate_concat<<<AA / 256, blk256, 0, stream>>>(attA, Vab, Wg, outA);
        k_gate_concat<<<AA / 256, blk256, 0, stream>>>(attB, Vbb, Wg, outB);
    }

    k_write_scalar<<<1, 32, 0, stream>>>(isz, out + (size_t)BB * 4 * CA);
}
